// MOEFFN_67482526154934
// MI455X (gfx1250) — matrix operand order repaired and checked by reading, not compiled, not run on hardware
//
#include <hip/hip_runtime.h>

#define EXP 8
#define DIM 1024
#define FDIM 2816
#define ROWS 16
#define WAVES 8

typedef __bf16 v16bf __attribute__((ext_vector_type(16)));
typedef float  v8f   __attribute__((ext_vector_type(8)));

union FragB {
    uint4 u[2];
    v16bf v;
};

static __device__ __forceinline__ unsigned short f2bf(float f) {
    unsigned int u = __float_as_uint(f);
    u += 0x7FFFu + ((u >> 16) & 1u);   // round-to-nearest-even
    return (unsigned short)(u >> 16);
}

static __device__ __forceinline__ v8f wmma_bf16(const FragB& a, const FragB& b, v8f c) {
    return __builtin_amdgcn_wmma_f32_16x16x32_bf16(
        /*neg_a=*/false, a.v, /*neg_b=*/false, b.v,
        /*c_mod=*/(short)0, c, /*reuse_a=*/false, /*reuse_b=*/false);
}

static __device__ __forceinline__ float fast_silu(float g) {
    // g * sigmoid(g) with v_exp_f32 + v_rcp_f32 (no IEEE div sequence)
    return g * __builtin_amdgcn_rcpf(1.f + __expf(-g));
}

// ---------------- fp32 -> bf16 conversion (vectorized) ----------------
__global__ void moe_cvt_bf16(const float* __restrict__ src,
                             unsigned short* __restrict__ dst, long n4) {
    long i = (long)blockIdx.x * blockDim.x + threadIdx.x;
    long stride = (long)gridDim.x * blockDim.x;
    for (; i < n4; i += stride) {
        float4 f = ((const float4*)src)[i];
        ushort4 o;
        o.x = f2bf(f.x); o.y = f2bf(f.y); o.z = f2bf(f.z); o.w = f2bf(f.w);
        ((ushort4*)dst)[i] = o;
    }
}

// ---------------- routing: scores, top-2, softmax, scatter ----------------
__global__ void moe_route(const float* __restrict__ x,
                          const float* __restrict__ gate_w,
                          int* __restrict__ cnt,
                          int* __restrict__ tok,
                          float* __restrict__ wtl,
                          int N) {
    __shared__ float sgw[EXP * DIM];        // 32 KB
    for (int i = threadIdx.x; i < EXP * DIM; i += blockDim.x) sgw[i] = gate_w[i];
    __syncthreads();

    int n = blockIdx.x * blockDim.x + threadIdx.x;
    if (n >= N) return;
    const float* xr = x + (size_t)n * DIM;

    float s[EXP];
#pragma unroll
    for (int e = 0; e < EXP; ++e) s[e] = 0.f;

    for (int d = 0; d < DIM; d += 4) {
        float4 xv = *(const float4*)(xr + d);
#pragma unroll
        for (int e = 0; e < EXP; ++e) {
            const float* g = sgw + e * DIM + d;
            s[e] += xv.x * g[0] + xv.y * g[1] + xv.z * g[2] + xv.w * g[3];
        }
    }

    float best = -3.4e38f, second = -3.4e38f;
    int bi = 0, si = 0;
#pragma unroll
    for (int e = 0; e < EXP; ++e) {
        float v = s[e];
        if (v > best)        { second = best; si = bi; best = v; bi = e; }
        else if (v > second) { second = v; si = e; }
    }
    float p0 = 1.f / (1.f + __expf(second - best));   // softmax over [best, second]
    float p1 = 1.f - p0;

    int pos0 = atomicAdd(&cnt[bi], 1);
    tok[bi * N + pos0] = n;  wtl[bi * N + pos0] = p0;
    int pos1 = atomicAdd(&cnt[si], 1);
    tok[si * N + pos1] = n;  wtl[si * N + pos1] = p1;
}

// ---------------- expert FFN: SwiGLU via WMMA bf16 ----------------
#define SM_A (ROWS * DIM * 2)            // 32 KB  gathered x tile (bf16)
#define SM_H (ROWS * FDIM * 2)           // 88 KB  h = silu(g)*u tile (bf16)
#define SMEM_BYTES (SM_A + SM_H + 2 * ROWS * 4 + 64)

__global__ void __launch_bounds__(256, 1)
moe_ffn(const unsigned short* __restrict__ x_bf,
        const unsigned short* __restrict__ wg_bf,
        const unsigned short* __restrict__ wu_bf,
        const unsigned short* __restrict__ wd_bf,
        const int*   __restrict__ tok,
        const float* __restrict__ wtl,
        const int*   __restrict__ cnt,
        float* __restrict__ out,
        int N) {
    extern __shared__ unsigned char smem[];
    unsigned short* Ah  = (unsigned short*)smem;                 // [ROWS][DIM]
    unsigned short* Hh  = (unsigned short*)(smem + SM_A);        // [ROWS][FDIM]
    int*   stok = (int*)  (smem + SM_A + SM_H);
    float* swt  = (float*)(smem + SM_A + SM_H + ROWS * 4);

    const int e  = blockIdx.y;
    const int count = cnt[e];
    const int r0 = blockIdx.x * ROWS;
    if (r0 >= count) return;

    if (threadIdx.x < ROWS) {
        int r = r0 + threadIdx.x;
        bool live = (r < count);
        stok[threadIdx.x] = live ? tok[e * N + r] : tok[e * N + r0];
        swt[threadIdx.x]  = live ? wtl[e * N + r] : 0.f;
    }
    __syncthreads();

    // gather x tile (ROWS rows of DIM bf16) into LDS, 16B chunks
    for (int c = threadIdx.x; c < ROWS * (DIM / 8); c += blockDim.x) {
        int row = c / (DIM / 8);
        int col = c % (DIM / 8);
        ((uint4*)Ah)[c] = ((const uint4*)(x_bf + (size_t)stok[row] * DIM))[col];
    }
    __syncthreads();

    const int wave  = threadIdx.x >> 5;
    const int lane  = threadIdx.x & 31;
    const int lhalf = lane >> 4;        // 0: lanes 0-15, 1: lanes 16-31
    const int l15   = lane & 15;

    // ---- phase 1: g = A * Wg^T, u = A * Wu^T, h = silu(g)*u -> LDS ----
    // Each wave owns a pair of adjacent 16-col F tiles: 1 A-frag feeds 4 WMMAs.
    const unsigned short* arow = Ah + l15 * DIM;
    for (int ftp = wave; ftp < FDIM / 32; ftp += WAVES) {
        const int f0 = ftp * 32;
        v8f g0 = {0.f,0.f,0.f,0.f,0.f,0.f,0.f,0.f};
        v8f u0 = {0.f,0.f,0.f,0.f,0.f,0.f,0.f,0.f};
        v8f g1 = {0.f,0.f,0.f,0.f,0.f,0.f,0.f,0.f};
        v8f u1 = {0.f,0.f,0.f,0.f,0.f,0.f,0.f,0.f};
        const unsigned short* wg0 = wg_bf + ((size_t)e * FDIM + f0 + l15) * DIM;
        const unsigned short* wu0 = wu_bf + ((size_t)e * FDIM + f0 + l15) * DIM;
        const unsigned short* wg1 = wg0 + 16 * DIM;
        const unsigned short* wu1 = wu0 + 16 * DIM;
        __builtin_prefetch(wg0, 0, 1);
        __builtin_prefetch(wu0, 0, 1);
#pragma unroll 2
        for (int k0 = 0; k0 < DIM; k0 += 32) {
            FragB a, bg0, bu0, bg1, bu1;
            const int ka = k0 + lhalf * 16;     // A layout: contiguous 16 K per lane-half
            a.u[0] = *(const uint4*)(arow + ka);
            a.u[1] = *(const uint4*)(arow + ka + 8);
            const int kb = k0 + lhalf * 16;     // B layout: contiguous 16 K per lane-half
            bg0.u[0] = *(const uint4*)(wg0 + kb);
            bg0.u[1] = *(const uint4*)(wg0 + kb + 8);
            bu0.u[0] = *(const uint4*)(wu0 + kb);
            bu0.u[1] = *(const uint4*)(wu0 + kb + 8);
            bg1.u[0] = *(const uint4*)(wg1 + kb);
            bg1.u[1] = *(const uint4*)(wg1 + kb + 8);
            bu1.u[0] = *(const uint4*)(wu1 + kb);
            bu1.u[1] = *(const uint4*)(wu1 + kb + 8);
            g0 = wmma_bf16(a, bg0, g0);
            u0 = wmma_bf16(a, bu0, u0);
            g1 = wmma_bf16(a, bg1, g1);
            u1 = wmma_bf16(a, bu1, u1);
        }
#pragma unroll
        for (int r = 0; r < 8; ++r) {
            int m = r + lhalf * 8;              // C/D layout: VGPR r -> row r (+8 hi lanes)
            Hh[m * FDIM + f0 + l15]      = f2bf(fast_silu(g0[r]) * u0[r]);
            Hh[m * FDIM + f0 + 16 + l15] = f2bf(fast_silu(g1[r]) * u1[r]);
        }
    }
    __syncthreads();

    // ---- phase 2: out_tile = h * Wd^T, scaled by routing weight, atomic combine ----
    const unsigned short* hrow = Hh + l15 * FDIM;
    for (int dtp = wave; dtp < DIM / 32; dtp += WAVES) {
        const int d0 = dtp * 32;
        v8f c0 = {0.f,0.f,0.f,0.f,0.f,0.f,0.f,0.f};
        v8f c1 = {0.f,0.f,0.f,0.f,0.f,0.f,0.f,0.f};
        const unsigned short* wd0 = wd_bf + ((size_t)e * DIM + d0 + l15) * FDIM;
        const unsigned short* wd1 = wd0 + 16 * FDIM;
        __builtin_prefetch(wd0, 0, 1);
#pragma unroll 2
        for (int k0 = 0; k0 < FDIM; k0 += 32) {
            FragB a, b0, b1;
            const int ka = k0 + lhalf * 16;
            a.u[0] = *(const uint4*)(hrow + ka);
            a.u[1] = *(const uint4*)(hrow + ka + 8);
            const int kb = k0 + lhalf * 16;
            b0.u[0] = *(const uint4*)(wd0 + kb);
            b0.u[1] = *(const uint4*)(wd0 + kb + 8);
            b1.u[0] = *(const uint4*)(wd1 + kb);
            b1.u[1] = *(const uint4*)(wd1 + kb + 8);
            c0 = wmma_bf16(a, b0, c0);
            c1 = wmma_bf16(a, b1, c1);
        }
#pragma unroll
        for (int r = 0; r < 8; ++r) {
            int m = r + lhalf * 8;
            float w = swt[m];
            float* orow = out + (size_t)stok[m] * DIM;
            __hip_atomic_fetch_add(orow + d0 + l15, w * c0[r],
                                   __ATOMIC_RELAXED, __HIP_MEMORY_SCOPE_AGENT);
            __hip_atomic_fetch_add(orow + d0 + 16 + l15, w * c1[r],
                                   __ATOMIC_RELAXED, __HIP_MEMORY_SCOPE_AGENT);
        }
    }
}

// ---------------- host launcher ----------------
extern "C" void kernel_launch(void* const* d_in, const int* in_sizes, int n_in,
                              void* d_out, int out_size, void* d_ws, size_t ws_size,
                              hipStream_t stream) {
    const float* x      = (const float*)d_in[0];
    const float* gate_w = (const float*)d_in[1];
    const float* w_gate = (const float*)d_in[2];
    const float* w_up   = (const float*)d_in[3];
    const float* w_down = (const float*)d_in[4];
    float* out = (float*)d_out;

    const int N = in_sizes[0] / DIM;                 // B*S = 4096
    const size_t WEX = (size_t)EXP * FDIM * DIM;     // elems per expert weight set

    unsigned char* ws = (unsigned char*)d_ws;
    size_t off = 0;
    int*   cnt  = (int*)(ws + off);          off += 256;
    int*   tok  = (int*)(ws + off);          off += (size_t)EXP * N * 4;
    float* wtl  = (float*)(ws + off);        off += (size_t)EXP * N * 4;
    unsigned short* x_bf  = (unsigned short*)(ws + off); off += (size_t)N * DIM * 2;
    unsigned short* wg_bf = (unsigned short*)(ws + off); off += WEX * 2;
    unsigned short* wu_bf = (unsigned short*)(ws + off); off += WEX * 2;
    unsigned short* wd_bf = (unsigned short*)(ws + off); off += WEX * 2;

    hipMemsetAsync(cnt, 0, 256, stream);
    hipMemsetAsync(out, 0, sizeof(float) * (size_t)out_size, stream);

    // bf16 conversions (weights become L2-resident: 139 MB < 192 MB L2)
    moe_cvt_bf16<<<2048, 256, 0, stream>>>(x,      x_bf,  (long)N * DIM / 4);
    moe_cvt_bf16<<<2048, 256, 0, stream>>>(w_gate, wg_bf, (long)WEX / 4);
    moe_cvt_bf16<<<2048, 256, 0, stream>>>(w_up,   wu_bf, (long)WEX / 4);
    moe_cvt_bf16<<<2048, 256, 0, stream>>>(w_down, wd_bf, (long)WEX / 4);

    moe_route<<<(N + 255) / 256, 256, 0, stream>>>(x, gate_w, cnt, tok, wtl, N);

    dim3 grid((N + ROWS - 1) / ROWS, EXP);
    moe_ffn<<<grid, 256, SMEM_BYTES, stream>>>(x_bf, wg_bf, wu_bf, wd_bf,
                                               tok, wtl, cnt, out, N);
}
